// NoisyGate_86517821214171
// MI455X (gfx1250) — matrix operand order repaired and checked by reading, not compiled, not run on hardware
//
#include <hip/hip_runtime.h>
#include <hip/hip_bf16.h>
#include <math.h>

// ---------------------------------------------------------------------------
// NoisyGate fused kernel for MI455X (gfx1250, wave32, WMMA)
//   clean = inp @ w_gate ; raw = inp @ w_noise   (fused: inp read ONCE)
//   GEMM via split-f16 (hi+lo) with v_wmma_f32_16x16x32_f16, f32 accumulate:
//   A*B ~= Ahi*Bhi + Ahi*Blo + Alo*Bhi  (~fp32 accuracy, 8x denser WMMA than
//   the f32 16x16x4 path -> matrix-op time drops from ~55us toward the ~11us
//   HBM floor). Gating epilogue fused (softplus, noisy top-3, softmax, cdf).
// ---------------------------------------------------------------------------

typedef _Float16 f16;
typedef __attribute__((ext_vector_type(4)))  _Float16 v4h;
typedef __attribute__((ext_vector_type(8)))  _Float16 v8h;
typedef __attribute__((ext_vector_type(16))) _Float16 v16h;
typedef __attribute__((ext_vector_type(8)))  float    v8f;

#define N_TOKENS   16384
#define D_MODEL    4096
#define N_EXPERT   64
#define K_CHUNK    32
#define KPAD       40      // 80B row pitch: 16B-aligned b128, bank-conflict-free
#define TOK_PER_BLK 128    // 8 waves * 16 tokens
#define GATES_OFF  (N_TOKENS * 2)   // 32768
#define LOSS_OFF   (N_TOKENS * 4)   // 65536

union SharedU {
    struct {
        _Float16 Ahi[8][16][KPAD];   // per-wave A tiles [wave][row][k]
        _Float16 Alo[8][16][KPAD];
        _Float16 Bhi[2][64][KPAD];   // transposed weights [mat][col][k]
        _Float16 Blo[2][64][KPAD];
    } s1;                            // 40 KB
    struct {
        float clean[8][16][64];
        float raw  [8][16][64];
    } s2;                            // 64 KB
};

static __device__ __forceinline__ v16h cat8(v8h a, v8h b) {
    return __builtin_shufflevector(a, b, 0,1,2,3,4,5,6,7,8,9,10,11,12,13,14,15);
}

__global__ void __launch_bounds__(256, 2)
noisy_gate_gemm_kernel(const float* __restrict__ inp,
                       const float* __restrict__ w_gate,
                       const float* __restrict__ w_noise,
                       const float* __restrict__ noise,
                       float* __restrict__ out,
                       float* __restrict__ ws)
{
    __shared__ SharedU sh;

    const int tid  = threadIdx.x;
    const int wave = tid >> 5;
    const int lane = tid & 31;
    const int tok0 = blockIdx.x * TOK_PER_BLK + wave * 16;

    v8f acc[2][4] = {};   // [gate|noise][n-tile] 16x16 f32 accumulators

    const int m   = lane & 15;   // row (A) / col-in-tile (B)
    const int sel = lane >> 4;   // half-wave K selector

    for (int kc = 0; kc < D_MODEL; kc += K_CHUNK) {
        // ---- stage A: each wave loads its 16 rows x 32 K, split into hi/lo f16
        {
            const int r  = lane >> 3;    // 0..3
            const int c4 = lane & 7;     // 0..7
#pragma unroll
            for (int p = 0; p < 4; ++p) {
                const int row = p * 4 + r;
                float4 v = *(const float4*)(inp + (size_t)(tok0 + row) * D_MODEL + kc + c4 * 4);
                f16 h0 = (f16)v.x, h1 = (f16)v.y, h2 = (f16)v.z, h3 = (f16)v.w;
                v4h hi = { h0, h1, h2, h3 };
                v4h lo = { (f16)(v.x - (float)h0), (f16)(v.y - (float)h1),
                           (f16)(v.z - (float)h2), (f16)(v.w - (float)h3) };
                *(v4h*)(&sh.s1.Ahi[wave][row][c4 * 4]) = hi;
                *(v4h*)(&sh.s1.Alo[wave][row][c4 * 4]) = lo;
            }
        }
        // ---- stage B (both matrices), split + transpose into [col][k]
#pragma unroll
        for (int q = 0; q < 4; ++q) {
            const int idx  = q * 256 + tid;    // 1024 float4 = 2 mats * 32 k * 16
            const int mat  = idx >> 9;
            const int rem  = idx & 511;
            const int krow = rem >> 4;         // 0..31
            const int c4   = rem & 15;         // 0..15
            const float* W = mat ? w_noise : w_gate;
            float4 v = *(const float4*)(W + (size_t)(kc + krow) * N_EXPERT + c4 * 4);
            const float fv[4] = { v.x, v.y, v.z, v.w };
#pragma unroll
            for (int j = 0; j < 4; ++j) {
                f16 h = (f16)fv[j];
                sh.s1.Bhi[mat][c4 * 4 + j][krow] = h;
                sh.s1.Blo[mat][c4 * 4 + j][krow] = (f16)(fv[j] - (float)h);
            }
        }
        __syncthreads();

        // prefetch next A chunk while the XDL pipe works (global_prefetch_b8)
        if (kc + K_CHUNK < D_MODEL) {
            __builtin_prefetch(inp + (size_t)(tok0 + m) * D_MODEL + kc + K_CHUNK, 0, 3);
        }

        // ---- A fragments: lanes 0-15 K{0-7,16-23}, lanes 16-31 K{8-15,24-31}
        v16h ah = cat8(*(const v8h*)(&sh.s1.Ahi[wave][m][sel * 8]),
                       *(const v8h*)(&sh.s1.Ahi[wave][m][16 + sel * 8]));
        v16h al = cat8(*(const v8h*)(&sh.s1.Alo[wave][m][sel * 8]),
                       *(const v8h*)(&sh.s1.Alo[wave][m][16 + sel * 8]));

        // ---- 3-product split GEMM: 2 mats x 4 n-tiles x 3 WMMA
#pragma unroll
        for (int mat = 0; mat < 2; ++mat) {
#pragma unroll
            for (int nt = 0; nt < 4; ++nt) {
                const int n = nt * 16 + m;
                // B fragments: lanes 0-15 K{0-7,16-23}, lanes 16-31 K{8-15,24-31}
                v16h bh = cat8(*(const v8h*)(&sh.s1.Bhi[mat][n][sel * 8]),
                               *(const v8h*)(&sh.s1.Bhi[mat][n][16 + sel * 8]));
                v16h bl = cat8(*(const v8h*)(&sh.s1.Blo[mat][n][sel * 8]),
                               *(const v8h*)(&sh.s1.Blo[mat][n][16 + sel * 8]));
                acc[mat][nt] = __builtin_amdgcn_wmma_f32_16x16x32_f16(
                    false, ah, false, bh, (short)0, acc[mat][nt], false, false);
                acc[mat][nt] = __builtin_amdgcn_wmma_f32_16x16x32_f16(
                    false, ah, false, bl, (short)0, acc[mat][nt], false, false);
                acc[mat][nt] = __builtin_amdgcn_wmma_f32_16x16x32_f16(
                    false, al, false, bh, (short)0, acc[mat][nt], false, false);
            }
        }
        __syncthreads();
    }

    // ---- spill accumulators to LDS (C layout: VGPR r -> M=r / r+8 by half-wave)
    {
#pragma unroll
        for (int nt = 0; nt < 4; ++nt) {
#pragma unroll
            for (int r = 0; r < 8; ++r) {
                const int row = r + sel * 8;
                sh.s2.clean[wave][row][nt * 16 + m] = acc[0][nt][r];
                sh.s2.raw  [wave][row][nt * 16 + m] = acc[1][nt][r];
            }
        }
    }
    __syncthreads();

    // ---- gating epilogue: one lane per token (lanes 0-15)
    float v0 = -1e30f, v1 = -1e30f, v2 = -1e30f;
    int   i0 = 0, i1 = 0, i2 = 0;
    const int t   = lane & 15;
    const int tok = tok0 + t;

    if (lane < 16) {
        for (int e = 0; e < N_EXPERT; ++e) {
            const float c  = sh.s2.clean[wave][t][e];
            const float r  = sh.s2.raw[wave][t][e];
            const float sp = (r > 20.0f) ? r : log1pf(__expf(r));
            const float sd = sp + 0.01f;
            const float nz = c + noise[(size_t)tok * N_EXPERT + e] * sd;
            if (nz > v0)      { v2 = v1; i2 = i1; v1 = v0; i1 = i0; v0 = nz; i0 = e; }
            else if (nz > v1) { v2 = v1; i2 = i1; v1 = nz; i1 = e; }
            else if (nz > v2) { v2 = nz; i2 = e; }
        }
        // softmax over top-2 (v0 >= v1)
        const float e1 = __expf(v1 - v0);
        const float g0 = 1.0f / (1.0f + e1);
        const float g1 = e1 / (1.0f + e1);
        out[(size_t)tok * 2 + 0] = (float)i0;
        out[(size_t)tok * 2 + 1] = (float)i1;
        out[GATES_OFF + (size_t)tok * 2 + 0] = g0;
        out[GATES_OFF + (size_t)tok * 2 + 1] = g1;
        atomicAdd(&ws[i0], g0);   // importance
        atomicAdd(&ws[i1], g1);
    }

    // ---- load = sum_t norm.cdf((clean - thr)/std); wave-reduce, 1 atomic/expert
    const float thrIn  = v2;   // 3rd largest (threshold_if_in)
    const float thrOut = v1;   // 2nd largest (threshold_if_out)
    for (int e = 0; e < N_EXPERT; ++e) {
        float p = 0.0f;
        if (lane < 16) {
            const float c  = sh.s2.clean[wave][t][e];
            const float r  = sh.s2.raw[wave][t][e];
            const float sp = (r > 20.0f) ? r : log1pf(__expf(r));
            const float sd = sp + 0.01f;
            const float nz = c + noise[(size_t)tok * N_EXPERT + e] * sd;
            const float th = (nz > thrIn) ? thrIn : thrOut;
            const float x  = (c - th) / sd;
            p = 0.5f * erfcf(-x * 0.70710678118654752f);   // norm.cdf
        }
#pragma unroll
        for (int off = 16; off > 0; off >>= 1) p += __shfl_xor(p, off, 32);
        if (lane == 0) atomicAdd(&ws[N_EXPERT + e], p);
    }
}

__global__ void noisy_gate_init_kernel(float* __restrict__ ws)
{
    const int i = threadIdx.x;
    if (i < 2 * N_EXPERT) ws[i] = 0.0f;
}

__global__ void noisy_gate_loss_kernel(const float* __restrict__ ws,
                                       float* __restrict__ out)
{
    __shared__ float si[N_EXPERT], sqi[N_EXPERT], sl[N_EXPERT], sql[N_EXPERT];
    const int e = threadIdx.x;
    const float imp = ws[e];
    const float ld  = ws[N_EXPERT + e];
    si[e]  = imp;  sqi[e] = imp * imp;
    sl[e]  = ld;   sql[e] = ld * ld;
    __syncthreads();
    if (e == 0) {
        float a = 0.f, b = 0.f, c = 0.f, d = 0.f;
        for (int i = 0; i < N_EXPERT; ++i) { a += si[i]; b += sqi[i]; c += sl[i]; d += sql[i]; }
        const float n  = (float)N_EXPERT;
        const float mi = a / n, ml = c / n;
        const float vi = (b - n * mi * mi) / (n - 1.0f);
        const float vl = (d - n * ml * ml) / (n - 1.0f);
        out[LOSS_OFF] = vi / (mi * mi + 1e-10f) + vl / (ml * ml + 1e-10f);
    }
}

extern "C" void kernel_launch(void* const* d_in, const int* in_sizes, int n_in,
                              void* d_out, int out_size, void* d_ws, size_t ws_size,
                              hipStream_t stream) {
    const float* inp     = (const float*)d_in[0];
    const float* w_gate  = (const float*)d_in[1];
    const float* w_noise = (const float*)d_in[2];
    const float* noise   = (const float*)d_in[3];
    float* out = (float*)d_out;
    float* ws  = (float*)d_ws;

    noisy_gate_init_kernel<<<1, 128, 0, stream>>>(ws);
    noisy_gate_gemm_kernel<<<N_TOKENS / TOK_PER_BLK, 256, 0, stream>>>(
        inp, w_gate, w_noise, noise, out, ws);
    noisy_gate_loss_kernel<<<1, N_EXPERT, 0, stream>>>(ws, out);
}
